// BiLSTM_CRF_80865644249886
// MI455X (gfx1250) — matrix operand order repaired and checked by reading, not compiled, not run on hardware
//
#include <hip/hip_runtime.h>
#include <hip/hip_bf16.h>
#include <math.h>

#define TLEN 4096
#define EDIM 256
#define HH   256
#define GDIM 1024   // 4*HH
#define HDIM 512
#define KTAG 32

typedef __attribute__((ext_vector_type(16))) __bf16 v16bf;
typedef __attribute__((ext_vector_type(8)))  float  v8f;

static __device__ __forceinline__ float bflo(unsigned u) { return __uint_as_float(u << 16); }
static __device__ __forceinline__ float bfhi(unsigned u) { return __uint_as_float(u & 0xffff0000u); }
static __device__ __forceinline__ float sigm(float x)    { return 1.0f / (1.0f + __expf(-x)); }

// ---------------- f32 -> bf16 weight conversion (Whh kept L2-resident in bf16) -------------
__global__ void cvt_bf16_kernel(const float* __restrict__ src, __bf16* __restrict__ dst, int n) {
  int i = blockIdx.x * blockDim.x + threadIdx.x;
  if (i < n) dst[i] = (__bf16)src[i];
}

// ---------------- Input projection: Xpre[t,j] = embed[sent[t']]·Wih[j,:] + bih[j]+bhh[j] ----
// One wave per 16x16 tile; v_wmma_f32_16x16x32_bf16, K = EDIM = 256 (8 k-steps, fully unrolled).
__global__ void input_proj_wmma(const int* __restrict__ sent,
                                const float* __restrict__ embed,
                                const float* __restrict__ Wih,   // (GDIM, EDIM) row-major
                                const float* __restrict__ bih,
                                const float* __restrict__ bhh,
                                float* __restrict__ Xpre,        // (TLEN, GDIM)
                                int rev) {
  const int NT = GDIM / 16;
  int gwave = (blockIdx.x * blockDim.x + threadIdx.x) >> 5;
  int lane  = threadIdx.x & 31;
  int mt = gwave / NT, nt = gwave % NT;
  int m0 = mt * 16, n0 = nt * 16;
  int half = lane >> 4;
  int l = lane & 15;

  int row  = m0 + l;
  int srow = rev ? (TLEN - 1 - row) : row;              // backward scan consumes x reversed
  const float* arow = embed + (size_t)sent[srow] * EDIM;
  const float* brow = Wih + (size_t)(n0 + l) * EDIM;

  int kbA = half * 16;   // A-frag: lanes0-15 K 0-15, lanes16-31 K 16-31
  int kbB = half * 16;   // B-frag (same order as A): lanes0-15 K 0-15, lanes16-31 K 16-31

  v8f acc = {};
#pragma unroll
  for (int k0 = 0; k0 < EDIM; k0 += 32) {
    v16bf a, b;
    const float* a0 = arow + k0 + kbA;
    const float* a1 = arow + k0 + kbA + 8;
#pragma unroll
    for (int i = 0; i < 8; ++i) { a[i] = (__bf16)a0[i]; a[8 + i] = (__bf16)a1[i]; }
    const float* bp = brow + k0 + kbB;
#pragma unroll
    for (int i = 0; i < 16; ++i) b[i] = (__bf16)bp[i];
    acc = __builtin_amdgcn_wmma_f32_16x16x32_bf16(false, a, false, b, (short)0, acc, false, false);
  }
  float bias = bih[n0 + l] + bhh[n0 + l];
#pragma unroll
  for (int r = 0; r < 8; ++r) {
    int m = m0 + half * 8 + r;                           // D layout: VGPR r -> M = r (+8 hi half)
    Xpre[(size_t)m * GDIM + n0 + l] = acc[r] + bias;
  }
}

// ---------------- Sequential LSTM recurrence: one WGP per direction -----------------------
// 1024 threads: thread j owns gate row j. Whh row (bf16, 512B) streamed from L2 each step,
// h broadcast from LDS. 4 independent accumulators keep the FMA chain short; next step's
// Xpre row is prefetched into the WGP cache while the dot product runs.
__global__ void lstm_rec_kernel(const float* __restrict__ Xpre_f,
                                const float* __restrict__ Xpre_b,
                                const __bf16* __restrict__ whh_f,
                                const __bf16* __restrict__ whh_b,
                                const float* __restrict__ h0,
                                const float* __restrict__ c0,
                                float* __restrict__ Hcat) {       // (TLEN, HDIM)
  __shared__ float sh[HH];
  __shared__ float sg[GDIM];
  int dir = blockIdx.x;
  int tid = threadIdx.x;
  const float*  Xpre = dir ? Xpre_b : Xpre_f;
  const __bf16* whh  = dir ? whh_b : whh_f;

  if (tid < HH) sh[tid] = h0[dir * HH + tid];
  float c = (tid < HH) ? c0[dir * HH + tid] : 0.0f;
  __syncthreads();

  const __bf16* wrow = whh + (size_t)tid * HH;
  for (int s = 0; s < TLEN; ++s) {
    if (s + 1 < TLEN)
      __builtin_prefetch(&Xpre[(size_t)(s + 1) * GDIM + tid], 0, 0);  // global_prefetch_b8

    float acc0 = 0.0f, acc1 = 0.0f, acc2 = 0.0f, acc3 = 0.0f;
#pragma unroll 4
    for (int k = 0; k < HH; k += 16) {
      uint4  w0 = *(const uint4*)(wrow + k);              // 8 bf16 weights (b128 from L2)
      uint4  w1 = *(const uint4*)(wrow + k + 8);
      float4 ha = *(const float4*)&sh[k];
      float4 hb = *(const float4*)&sh[k + 4];
      float4 hc = *(const float4*)&sh[k + 8];
      float4 hd = *(const float4*)&sh[k + 12];
      acc0 = fmaf(bflo(w0.x), ha.x, acc0);
      acc1 = fmaf(bfhi(w0.x), ha.y, acc1);
      acc2 = fmaf(bflo(w0.y), ha.z, acc2);
      acc3 = fmaf(bfhi(w0.y), ha.w, acc3);
      acc0 = fmaf(bflo(w0.z), hb.x, acc0);
      acc1 = fmaf(bfhi(w0.z), hb.y, acc1);
      acc2 = fmaf(bflo(w0.w), hb.z, acc2);
      acc3 = fmaf(bfhi(w0.w), hb.w, acc3);
      acc0 = fmaf(bflo(w1.x), hc.x, acc0);
      acc1 = fmaf(bfhi(w1.x), hc.y, acc1);
      acc2 = fmaf(bflo(w1.y), hc.z, acc2);
      acc3 = fmaf(bfhi(w1.y), hc.w, acc3);
      acc0 = fmaf(bflo(w1.z), hd.x, acc0);
      acc1 = fmaf(bfhi(w1.z), hd.y, acc1);
      acc2 = fmaf(bflo(w1.w), hd.z, acc2);
      acc3 = fmaf(bfhi(w1.w), hd.w, acc3);
    }
    sg[tid] = ((acc0 + acc1) + (acc2 + acc3)) + Xpre[(size_t)s * GDIM + tid];
    __syncthreads();
    if (tid < HH) {
      float gi = sg[tid], gf = sg[HH + tid], gg = sg[2 * HH + tid], go = sg[3 * HH + tid];
      c = sigm(gf) * c + sigm(gi) * tanhf(gg);
      float h = sigm(go) * tanhf(c);
      sh[tid] = h;
      int t = dir ? (TLEN - 1 - s) : s;
      Hcat[(size_t)t * HDIM + dir * HH + tid] = h;
    }
    __syncthreads();
  }
}

// ---------------- Output projection: feats = Hcat @ W_out.T + b_out (WMMA, N=32) ----------
__global__ void out_proj_wmma(const float* __restrict__ Hcat,    // (TLEN, HDIM)
                              const float* __restrict__ W_out,   // (KTAG, HDIM)
                              const float* __restrict__ b_out,
                              float* __restrict__ feats) {       // (TLEN, KTAG)
  const int NT = KTAG / 16;
  int gwave = (blockIdx.x * blockDim.x + threadIdx.x) >> 5;
  int lane  = threadIdx.x & 31;
  int mt = gwave / NT, nt = gwave % NT;
  int m0 = mt * 16, n0 = nt * 16;
  int half = lane >> 4;
  int l = lane & 15;

  const float* arow = Hcat + (size_t)(m0 + l) * HDIM;
  const float* brow = W_out + (size_t)(n0 + l) * HDIM;
  int kbA = half * 16, kbB = half * 16;

  v8f acc = {};
#pragma unroll
  for (int k0 = 0; k0 < HDIM; k0 += 32) {
    v16bf a, b;
    const float* a0 = arow + k0 + kbA;
    const float* a1 = arow + k0 + kbA + 8;
#pragma unroll
    for (int i = 0; i < 8; ++i) { a[i] = (__bf16)a0[i]; a[8 + i] = (__bf16)a1[i]; }
    const float* bp = brow + k0 + kbB;
#pragma unroll
    for (int i = 0; i < 16; ++i) b[i] = (__bf16)bp[i];
    acc = __builtin_amdgcn_wmma_f32_16x16x32_bf16(false, a, false, b, (short)0, acc, false, false);
  }
  float bias = b_out[n0 + l];
#pragma unroll
  for (int r = 0; r < 8; ++r) {
    int m = m0 + half * 8 + r;
    feats[(size_t)m * KTAG + n0 + l] = acc[r] + bias;
  }
}

// ---------------- Viterbi: one wave32; trans column in VGPRs, backpointers in LDS ---------
__global__ void viterbi_kernel(const float* __restrict__ feats,
                               const float* __restrict__ trans,  // (KTAG, KTAG)
                               float* __restrict__ out) {        // out[0..T-1]=path, out[T]=score
  extern __shared__ unsigned char smem[];
  unsigned char* bp   = smem;                       // TLEN*KTAG bytes (128 KB of the 320 KB LDS)
  float*         prev = (float*)(smem + TLEN * KTAG);
  int j = threadIdx.x;                              // lane = tag
  float tr[KTAG];
#pragma unroll
  for (int i = 0; i < KTAG; ++i) tr[i] = trans[i * KTAG + j];
  prev[j] = feats[j];
  __syncthreads();

  for (int t = 1; t < TLEN; ++t) {
    float obs = feats[t * KTAG + j];
    float best = -3.4e38f; int bix = 0;
#pragma unroll
    for (int i = 0; i < KTAG; ++i) {
      float sc = prev[i] + tr[i];
      if (sc > best) { best = sc; bix = i; }        // first-max tie-break (matches argmax)
    }
    __syncthreads();
    prev[j] = best + obs;
    bp[t * KTAG + j] = (unsigned char)bix;
    __syncthreads();
  }

  if (j == 0) {
    float score = prev[0]; int cur = 0;
    for (int i = 1; i < KTAG; ++i) if (prev[i] > score) { score = prev[i]; cur = i; }
    out[TLEN] = score;
    out[TLEN - 1] = (float)cur;
    for (int t = TLEN - 1; t >= 1; --t) {           // backtrace entirely in LDS
      cur = bp[t * KTAG + cur];
      out[t - 1] = (float)cur;
    }
  }
}

extern "C" void kernel_launch(void* const* d_in, const int* in_sizes, int n_in,
                              void* d_out, int out_size, void* d_ws, size_t ws_size,
                              hipStream_t stream) {
  const int*   sent  = (const int*)d_in[0];
  const float* embed = (const float*)d_in[1];
  const float* Wih_f = (const float*)d_in[2];
  const float* Whh_f = (const float*)d_in[3];
  const float* bih_f = (const float*)d_in[4];
  const float* bhh_f = (const float*)d_in[5];
  const float* Wih_b = (const float*)d_in[6];
  const float* Whh_b = (const float*)d_in[7];
  const float* bih_b = (const float*)d_in[8];
  const float* bhh_b = (const float*)d_in[9];
  const float* h0    = (const float*)d_in[10];
  const float* c0    = (const float*)d_in[11];
  const float* W_out = (const float*)d_in[12];
  const float* b_out = (const float*)d_in[13];
  const float* trans = (const float*)d_in[14];
  float* out = (float*)d_out;

  char* ws = (char*)d_ws;
  float*  Xpre_f = (float*)(ws);                                   // 16 MB
  float*  Xpre_b = (float*)(ws + (size_t)16 * 1024 * 1024);        // 16 MB
  float*  Hcat   = (float*)(ws + (size_t)32 * 1024 * 1024);        // 8 MB
  float*  feats  = (float*)(ws + (size_t)40 * 1024 * 1024);        // 512 KB
  __bf16* whh_fb = (__bf16*)(ws + (size_t)41 * 1024 * 1024);       // 512 KB
  __bf16* whh_bb = (__bf16*)(ws + (size_t)41 * 1024 * 1024 + 512 * 1024);

  const int nW = GDIM * HH;
  cvt_bf16_kernel<<<(nW + 255) / 256, 256, 0, stream>>>(Whh_f, whh_fb, nW);
  cvt_bf16_kernel<<<(nW + 255) / 256, 256, 0, stream>>>(Whh_b, whh_bb, nW);

  const int tilesA = (TLEN / 16) * (GDIM / 16);   // 16384 waves, 8 waves/block
  input_proj_wmma<<<tilesA / 8, 256, 0, stream>>>(sent, embed, Wih_f, bih_f, bhh_f, Xpre_f, 0);
  input_proj_wmma<<<tilesA / 8, 256, 0, stream>>>(sent, embed, Wih_b, bih_b, bhh_b, Xpre_b, 1);

  lstm_rec_kernel<<<2, 1024, 0, stream>>>(Xpre_f, Xpre_b, whh_fb, whh_bb, h0, c0, Hcat);

  const int tilesC = (TLEN / 16) * (KTAG / 16);   // 512 waves
  out_proj_wmma<<<tilesC / 8, 256, 0, stream>>>(Hcat, W_out, b_out, feats);

  viterbi_kernel<<<1, KTAG, TLEN * KTAG + KTAG * (int)sizeof(float), stream>>>(feats, trans, out);
}
